// DFlashAttention_32289564131757
// MI455X (gfx1250) — matrix operand order repaired and checked by reading, not compiled, not run on hardware
//
#include <hip/hip_runtime.h>
#include <hip/hip_bf16.h>

// ---------------------------------------------------------------------------
// Types for CDNA5 WMMA (wave32): v_wmma_f32_16x16x32_bf16
// ---------------------------------------------------------------------------
typedef __attribute__((ext_vector_type(16))) __bf16 v16bf;
typedef __attribute__((ext_vector_type(8)))  __bf16 v8bf;
typedef __attribute__((ext_vector_type(8)))  float  v8f;

union Frag16 { v16bf v; v8bf h[2]; };

__device__ __forceinline__ v8f wmma_bf16(v16bf a, v16bf b, v8f c) {
  // (neg_a, A, neg_b, B, c_mod, C, reuse_a, reuse_b)
  return __builtin_amdgcn_wmma_f32_16x16x32_bf16(false, a, false, b, (short)0, c,
                                                 false, false);
}

// Problem constants
#define QLEN   1024
#define CTXLEN 2048
#define TLEN   3072
#define HID    2048
#define NQH    32
#define NKV    8
#define HD     128

// ---------------------------------------------------------------------------
// K0: f32 -> bf16 convert (grid-stride)
// ---------------------------------------------------------------------------
__global__ void f32_to_bf16_kernel(const float* __restrict__ in,
                                   __bf16* __restrict__ out, long n) {
  long i = (long)blockIdx.x * blockDim.x + threadIdx.x;
  long stride = (long)gridDim.x * blockDim.x;
  for (; i < n; i += stride) out[i] = (__bf16)in[i];
}

// ---------------------------------------------------------------------------
// K1: GEMM  C[M,N] = A[M,K] * W[N,K]^T   (A,W bf16 row-major, C f32)
// Block 256 threads = 8 wave32 waves; tile 128(M) x 128(N); wave = 32x64.
// Double-buffered fragments: iteration k's loads overlap iteration k-1's
// WMMAs, so waits become partial (loadcnt<=8) instead of loadcnt==0.
// A-fragment (16x32 bf16): lane m=l&15 holds 16 contiguous K at hi*16.
// B-fragment (32x16 bf16): lane col n=l&15 holds 16 contiguous K at hi*16.
// C-fragment: row = r + 8*(l>>4), col = l&15.
// ---------------------------------------------------------------------------
__global__ __launch_bounds__(256) void gemm_bf16_nt(
    const __bf16* __restrict__ A, const __bf16* __restrict__ W,
    float* __restrict__ C, int M, int N, int K) {
  const int wid  = threadIdx.x >> 5;
  const int lane = threadIdx.x & 31;
  const int lo   = lane & 15;
  const int hi   = lane >> 4;
  const int wm   = wid & 3;                 // 4 waves along M
  const int wn   = wid >> 2;                // 2 waves along N
  const int mbase = blockIdx.y * 128 + wm * 32;
  const int nbase = blockIdx.x * 128 + wn * 64;

  v8f acc[2][4];
#pragma unroll
  for (int i = 0; i < 2; ++i)
#pragma unroll
    for (int j = 0; j < 4; ++j) acc[i][j] = (v8f){0.f,0.f,0.f,0.f,0.f,0.f,0.f,0.f};

  const __bf16* arow0 = A + (size_t)(mbase + lo) * K;
  const __bf16* arow1 = A + (size_t)(mbase + 16 + lo) * K;
  const __bf16* wrow[4];
#pragma unroll
  for (int j = 0; j < 4; ++j) wrow[j] = W + (size_t)(nbase + 16 * j + lo) * K;

  Frag16 a0[2], a1[2];
  v16bf  b[2][4];

  // prologue: fill buffer 0 with k0 = 0
  {
    const int ka = hi * 16;
    const int kb = hi * 16;
    a0[0].h[0] = *(const v8bf*)(arow0 + ka);
    a0[0].h[1] = *(const v8bf*)(arow0 + ka + 8);
    a1[0].h[0] = *(const v8bf*)(arow1 + ka);
    a1[0].h[1] = *(const v8bf*)(arow1 + ka + 8);
#pragma unroll
    for (int j = 0; j < 4; ++j) b[0][j] = *(const v16bf*)(wrow[j] + kb);
  }

  int buf = 0;
  for (int k0 = 0; k0 < K; k0 += 32) {
    const int nxt = buf ^ 1;
    if (k0 + 32 < K) {
      const int ka = k0 + 32 + hi * 16;
      const int kb = k0 + 32 + hi * 16;
      __builtin_prefetch(arow0 + k0 + 160, 0, 0);   // global_prefetch_b8
      __builtin_prefetch(wrow[0] + k0 + 160, 0, 0);
      a0[nxt].h[0] = *(const v8bf*)(arow0 + ka);
      a0[nxt].h[1] = *(const v8bf*)(arow0 + ka + 8);
      a1[nxt].h[0] = *(const v8bf*)(arow1 + ka);
      a1[nxt].h[1] = *(const v8bf*)(arow1 + ka + 8);
#pragma unroll
      for (int j = 0; j < 4; ++j) b[nxt][j] = *(const v16bf*)(wrow[j] + kb);
    }
#pragma unroll
    for (int j = 0; j < 4; ++j) {
      acc[0][j] = wmma_bf16(a0[buf].v, b[buf][j], acc[0][j]);
      acc[1][j] = wmma_bf16(a1[buf].v, b[buf][j], acc[1][j]);
    }
    buf = nxt;
  }

#pragma unroll
  for (int i = 0; i < 2; ++i)
#pragma unroll
    for (int j = 0; j < 4; ++j) {
      float* cp = C + (size_t)(mbase + 16 * i + 8 * hi) * N + nbase + 16 * j + lo;
#pragma unroll
      for (int r = 0; r < 8; ++r) cp[(size_t)r * N] = acc[i][j][r];
    }
}

// ---------------------------------------------------------------------------
// K2: per-head RMS norm + RoPE; f32 proj -> bf16 head-major.
// One wave per 128-dim vector; lane owns d = {lane, lane+32, lane+64, lane+96}
// so rotate_half partners (d <-> d^64) stay in-lane.
// ---------------------------------------------------------------------------
__global__ __launch_bounds__(256) void qk_postprocess(
    const float* __restrict__ proj, const float* __restrict__ nw,
    const float* __restrict__ cosT, const float* __restrict__ sinT,
    __bf16* __restrict__ outb, int nvec, int heads, int seq, int posoff,
    int rowstride) {
  const int wid  = threadIdx.x >> 5;
  const int lane = threadIdx.x & 31;
  const int v = blockIdx.x * 8 + wid;
  if (v >= nvec) return;
  const int row = v / heads;
  const int h   = v % heads;

  const float* src = proj + (size_t)row * rowstride + h * HD;
  float x[4];
#pragma unroll
  for (int i = 0; i < 4; ++i) x[i] = src[lane + 32 * i];

  float ss = x[0]*x[0] + x[1]*x[1] + x[2]*x[2] + x[3]*x[3];
#pragma unroll
  for (int off = 1; off < 32; off <<= 1) ss += __shfl_xor(ss, off, 32);
  const float inv = rsqrtf(ss * (1.0f / 128.0f) + 1e-6f);

  float xn[4];
#pragma unroll
  for (int i = 0; i < 4; ++i) xn[i] = x[i] * inv * nw[lane + 32 * i];

  const int pos = posoff + row;
  const float* cp = cosT + (size_t)pos * HD;
  const float* sp = sinT + (size_t)pos * HD;
  float c[4], s[4];
#pragma unroll
  for (int i = 0; i < 4; ++i) { c[i] = cp[lane + 32*i]; s[i] = sp[lane + 32*i]; }

  float y[4];
  y[0] = xn[0] * c[0] - xn[2] * s[0];
  y[1] = xn[1] * c[1] - xn[3] * s[1];
  y[2] = xn[2] * c[2] + xn[0] * s[2];
  y[3] = xn[3] * c[3] + xn[1] * s[3];

  __bf16* dst = outb + ((size_t)h * seq + row) * HD;
#pragma unroll
  for (int i = 0; i < 4; ++i) dst[lane + 32 * i] = (__bf16)y[i];
}

// ---------------------------------------------------------------------------
// K3: V f32 [T, NKV*HD] -> bf16 transposed [NKV, HD, T]
// ---------------------------------------------------------------------------
__global__ void v_transpose_kernel(const float* __restrict__ V,
                                   __bf16* __restrict__ Vt) {
  long idx = (long)blockIdx.x * blockDim.x + threadIdx.x;
  if (idx >= (long)TLEN * NKV * HD) return;
  const int t = (int)(idx >> 10);       // / (NKV*HD) = /1024
  const int c = (int)(idx & 1023);
  const int h = c >> 7;
  const int d = c & 127;
  Vt[((size_t)h * HD + d) * TLEN + t] = (__bf16)V[idx];
}

// ---------------------------------------------------------------------------
// K4: Flash attention. 8 waves/block, one wave per (head, 16-row q-tile).
// Streams T in 32-col chunks: 8 score WMMAs + online softmax + LDS transpose
// of P into A-fragment layout + 8 PV WMMAs. Wave-private LDS (in-order DS
// within a wave) => no block barrier, so each wave exits at its causal bound.
// ---------------------------------------------------------------------------
__global__ __launch_bounds__(256) void attn_kernel(
    const __bf16* __restrict__ Qb,   // [NQH, QLEN, HD]
    const __bf16* __restrict__ Kb,   // [NKV, TLEN, HD]
    const __bf16* __restrict__ Vt,   // [NKV, HD, TLEN]
    __bf16* __restrict__ Obf) {      // [QLEN, NQH*HD]
  __shared__ __align__(16) __bf16 pbuf[8][16 * 40];  // 40-elem row stride (80B)

  const int wid  = threadIdx.x >> 5;
  const int lane = threadIdx.x & 31;
  const int lo   = lane & 15;
  const int hi   = lane >> 4;
  const int gw   = blockIdx.x * 8 + wid;    // 0..2047
  const int head = gw >> 6;                 // 32 heads
  const int qt   = gw & 63;                 // 64 q-tiles of 16
  const int qbase = qt * 16;
  const int kvh  = head >> 2;               // n_rep = 4

  // Preload Q A-fragments for all 4 d-chunks of 32
  const __bf16* qrow = Qb + ((size_t)head * QLEN + qbase + lo) * HD;
  Frag16 qa[4];
#pragma unroll
  for (int cchunk = 0; cchunk < 4; ++cchunk) {
    qa[cchunk].h[0] = *(const v8bf*)(qrow + 32 * cchunk + hi * 16);
    qa[cchunk].h[1] = *(const v8bf*)(qrow + 32 * cchunk + hi * 16 + 8);
  }

  v8f acco[8];
#pragma unroll
  for (int dt = 0; dt < 8; ++dt) acco[dt] = (v8f){0.f,0.f,0.f,0.f,0.f,0.f,0.f,0.f};
  float rowm[8], rowl[8];
#pragma unroll
  for (int r = 0; r < 8; ++r) { rowm[r] = -3.0e38f; rowl[r] = 0.0f; }

  const float scale = 0.08838834764831845f;   // 128^-0.5
  const int tEnd = CTXLEN + qbase + 16;       // causal bound (exclusive)

  for (int t0 = 0; t0 < tEnd; t0 += 32) {
    // ---- scores: two 16x16 tiles over cols [t0, t0+32) ----
    v8f sc[2];
#pragma unroll
    for (int j = 0; j < 2; ++j) {
      sc[j] = (v8f){0.f,0.f,0.f,0.f,0.f,0.f,0.f,0.f};
      const int t = t0 + 16 * j + lo;
      const __bf16* krow = Kb + ((size_t)kvh * TLEN + t) * HD;
#pragma unroll
      for (int cchunk = 0; cchunk < 4; ++cchunk) {
        v16bf kb = *(const v16bf*)(krow + 32 * cchunk + hi * 16);
        sc[j] = wmma_bf16(qa[cchunk].v, kb, sc[j]);
      }
    }

    // ---- mask + online softmax (rows r + 8*hi, col = lo + 16*j) ----
#pragma unroll
    for (int r = 0; r < 8; ++r) {
      const int row = qbase + 8 * hi + r;
#pragma unroll
      for (int j = 0; j < 2; ++j) {
        const int t = t0 + 16 * j + lo;
        const bool ok = (t < CTXLEN) || ((t - CTXLEN) <= row);
        sc[j][r] = ok ? sc[j][r] * scale : -1.0e30f;
      }
      float cmax = fmaxf(sc[0][r], sc[1][r]);
#pragma unroll
      for (int off = 1; off < 16; off <<= 1)
        cmax = fmaxf(cmax, __shfl_xor(cmax, off, 32));
      const float nm   = fmaxf(rowm[r], cmax);
      const float corr = __expf(rowm[r] - nm);
      rowm[r] = nm;
      const float p0 = __expf(sc[0][r] - nm);
      const float p1 = __expf(sc[1][r] - nm);
      float rs = p0 + p1;
#pragma unroll
      for (int off = 1; off < 16; off <<= 1) rs += __shfl_xor(rs, off, 32);
      rowl[r] = rowl[r] * corr + rs;
#pragma unroll
      for (int dt = 0; dt < 8; ++dt) acco[dt][r] = acco[dt][r] * corr;

      const int rr = 8 * hi + r;
      pbuf[wid][rr * 40 + lo]      = (__bf16)p0;
      pbuf[wid][rr * 40 + 16 + lo] = (__bf16)p1;
    }
    __asm__ volatile("" ::: "memory");   // keep DS write->read order (in-order per wave)

    // ---- P as A-fragment (16 rows x 32 t-cols) ----
    Frag16 pa;
    pa.h[0] = *(const v8bf*)(&pbuf[wid][lo * 40 + hi * 16]);
    pa.h[1] = *(const v8bf*)(&pbuf[wid][lo * 40 + hi * 16 + 8]);
    __asm__ volatile("" ::: "memory");

    // ---- out += P @ V  (8 d-tiles of 16) ----
#pragma unroll
    for (int dt = 0; dt < 8; ++dt) {
      const int d = 16 * dt + lo;
      const __bf16* vrow = Vt + ((size_t)kvh * HD + d) * TLEN + t0 + hi * 16;
      v16bf vb = *(const v16bf*)vrow;
      acco[dt] = wmma_bf16(pa.v, vb, acco[dt]);
    }
  }

  // ---- finalize: divide by row sums, store bf16 to [q, head*128 + d] ----
#pragma unroll
  for (int r = 0; r < 8; ++r) {
    const int row = qbase + 8 * hi + r;
    const float invl = (rowl[r] > 0.0f) ? (1.0f / rowl[r]) : 0.0f;
#pragma unroll
    for (int dt = 0; dt < 8; ++dt) {
      Obf[(size_t)row * (NQH * HD) + head * HD + 16 * dt + lo] =
          (__bf16)(acco[dt][r] * invl);
    }
  }
}

// ---------------------------------------------------------------------------
// Host-side launch
// ---------------------------------------------------------------------------
extern "C" void kernel_launch(void* const* d_in, const int* in_sizes, int n_in,
                              void* d_out, int out_size, void* d_ws, size_t ws_size,
                              hipStream_t stream) {
  const float* hidden = (const float*)d_in[0];   // [1,1024,2048]
  const float* target = (const float*)d_in[1];   // [1,2048,2048]
  const float* cosT   = (const float*)d_in[2];   // [1,3072,128]
  const float* sinT   = (const float*)d_in[3];   // [1,3072,128]
  // d_in[4] attention_mask: reconstructed analytically in-kernel
  const float* wq     = (const float*)d_in[5];   // [4096,2048]
  const float* wk     = (const float*)d_in[6];   // [1024,2048]
  const float* wv     = (const float*)d_in[7];   // [1024,2048]
  const float* wo     = (const float*)d_in[8];   // [2048,4096]
  const float* qnw    = (const float*)d_in[9];   // [128]
  const float* knw    = (const float*)d_in[10];  // [128]
  float* out = (float*)d_out;                    // [1024,2048]

  char* ws = (char*)d_ws;
  size_t off = 0;
  auto alloc = [&](size_t bytes) -> void* {
    void* p = ws + off;
    off = (off + bytes + 255) & ~(size_t)255;
    return p;
  };

  __bf16* hidden_bf = (__bf16*)alloc((size_t)QLEN * HID * 2);
  __bf16* target_bf = (__bf16*)alloc((size_t)CTXLEN * HID * 2);
  __bf16* wq_bf     = (__bf16*)alloc((size_t)NQH * HD * HID * 2);
  __bf16* wk_bf     = (__bf16*)alloc((size_t)NKV * HD * HID * 2);
  __bf16* wv_bf     = (__bf16*)alloc((size_t)NKV * HD * HID * 2);
  __bf16* wo_bf     = (__bf16*)alloc((size_t)HID * NQH * HD * 2);
  float*  Qproj     = (float*)alloc((size_t)QLEN * NQH * HD * 4);
  float*  Kproj     = (float*)alloc((size_t)TLEN * NKV * HD * 4);
  float*  Vproj     = (float*)alloc((size_t)TLEN * NKV * HD * 4);
  __bf16* Qb        = (__bf16*)alloc((size_t)NQH * QLEN * HD * 2);
  __bf16* Kb        = (__bf16*)alloc((size_t)NKV * TLEN * HD * 2);
  __bf16* Vt        = (__bf16*)alloc((size_t)NKV * HD * TLEN * 2);
  __bf16* Obf       = (__bf16*)alloc((size_t)QLEN * NQH * HD * 2);

  auto conv = [&](const float* src, __bf16* dst, long n) {
    int blocks = (int)((n + 255) / 256);
    if (blocks > 4096) blocks = 4096;
    f32_to_bf16_kernel<<<blocks, 256, 0, stream>>>(src, dst, n);
  };
  conv(hidden, hidden_bf, (long)QLEN * HID);
  conv(target, target_bf, (long)CTXLEN * HID);
  conv(wq, wq_bf, (long)NQH * HD * HID);
  conv(wk, wk_bf, (long)NKV * HD * HID);
  conv(wv, wv_bf, (long)NKV * HD * HID);
  conv(wo, wo_bf, (long)HID * NQH * HD);

  auto gemm = [&](const __bf16* A, const __bf16* W, float* C, int M, int N, int K) {
    dim3 grid(N / 128, M / 128);
    gemm_bf16_nt<<<grid, 256, 0, stream>>>(A, W, C, M, N, K);
  };
  // Q projection: [1024, 4096]
  gemm(hidden_bf, wq_bf, Qproj, QLEN, NQH * HD, HID);
  // K projection: rows [0,2048) from target, [2048,3072) from hidden
  gemm(target_bf, wk_bf, Kproj, CTXLEN, NKV * HD, HID);
  gemm(hidden_bf, wk_bf, Kproj + (size_t)CTXLEN * NKV * HD, QLEN, NKV * HD, HID);
  // V projection
  gemm(target_bf, wv_bf, Vproj, CTXLEN, NKV * HD, HID);
  gemm(hidden_bf, wv_bf, Vproj + (size_t)CTXLEN * NKV * HD, QLEN, NKV * HD, HID);

  // RMS norm + RoPE -> head-major bf16
  qk_postprocess<<<(QLEN * NQH) / 8, 256, 0, stream>>>(
      Qproj, qnw, cosT, sinT, Qb, QLEN * NQH, NQH, QLEN, CTXLEN, NQH * HD);
  qk_postprocess<<<(TLEN * NKV) / 8, 256, 0, stream>>>(
      Kproj, knw, cosT, sinT, Kb, TLEN * NKV, NKV, TLEN, 0, NKV * HD);

  // V transpose -> [NKV, HD, TLEN] bf16
  v_transpose_kernel<<<((long)TLEN * NKV * HD + 255) / 256, 256, 0, stream>>>(Vproj, Vt);

  // Flash attention: 2048 (head, qtile) waves, 8 per block
  attn_kernel<<<(NQH * (QLEN / 16)) / 8, 256, 0, stream>>>(Qb, Kb, Vt, Obf);

  // Output projection: [1024, 2048] f32 into d_out
  gemm(Obf, wo_bf, out, QLEN, HID, NQH * HD);
}
